// ProductOfGaussiansMultimodal_721554506120
// MI455X (gfx1250) — matrix operand order repaired and checked by reading, not compiled, not run on hardware
//
#include <hip/hip_runtime.h>
#include <hip/hip_bf16.h>
#include <math.h>

#define T_STEPS 4096
#define DXN 128
#define DCN 64
#define DZN 32
#define DHN 1024

#define TM 128
#define TN 128
#define TK 32

typedef __attribute__((ext_vector_type(16))) __bf16        v16bf;
typedef __attribute__((ext_vector_type(8)))  float         v8f;
typedef __attribute__((ext_vector_type(8)))  unsigned int  v8u;

__device__ __forceinline__ unsigned int pack2_bf16_rne(float a, float b) {
  unsigned int ua = __builtin_bit_cast(unsigned int, a);
  unsigned int ub = __builtin_bit_cast(unsigned int, b);
  ua += 0x7FFFu + ((ua >> 16) & 1u);
  ub += 0x7FFFu + ((ub >> 16) & 1u);
  return (ua >> 16) | (ub & 0xFFFF0000u);
}

// ---------------------------------------------------------------------------
// Y[M,N] = act( X[M,K] @ W[N,K]^T + bias ), X optionally split at k=ksplit
// between X0 and X1 (concat layer). bf16 WMMA, f32 accumulate.
// Block: 256 threads (8 waves). Tile: 128x128, K-chunk 32, double-buffered LDS.
// Each wave owns a 32x64 slice: 2 A-frags x 4 B-frags = 8 WMMA per K-chunk.
// ---------------------------------------------------------------------------
__global__ __launch_bounds__(256) void gemm_bias_act_wmma(
    const float* __restrict__ X0, const float* __restrict__ X1, int ksplit,
    const float* __restrict__ W, const float* __restrict__ bias,
    float* __restrict__ Y, int M, int N, int K, int relu)
{
  __shared__ __attribute__((aligned(16))) unsigned short As[2][TM][TK];
  __shared__ __attribute__((aligned(16))) unsigned short Bs[2][TN][TK];

  const int tid   = threadIdx.x;
  const int lane  = tid & 31;
  const int wave  = tid >> 5;
  const int wm    = wave & 3;    // M group (32 rows each)
  const int wn    = wave >> 2;   // N group (64 cols each)
  const int tileM = blockIdx.y * TM;
  const int tileN = blockIdx.x * TN;

  v8f acc[2][4];
  #pragma unroll
  for (int a = 0; a < 2; ++a)
    #pragma unroll
    for (int n = 0; n < 4; ++n)
      #pragma unroll
      for (int j = 0; j < 8; ++j) acc[a][n][j] = 0.0f;

  // stage one 128x32 A tile + 128x32 B tile (f32 -> packed bf16) into buf
  auto stage = [&](int k0, int buf) {
    // A: ksplit is a multiple of 32, so a float4 never crosses the X0/X1 seam
    for (int idx = tid * 4; idx < TM * TK; idx += 256 * 4) {
      const int r = idx / TK, cc = idx % TK;
      const int gr = tileM + r, gc = k0 + cc;
      const float* src = (gc < ksplit)
          ? (X0 + (size_t)gr * (size_t)ksplit + gc)
          : (X1 + (size_t)gr * (size_t)(K - ksplit) + (gc - ksplit));
      const float4 v = *(const float4*)src;
      uint2 p;
      p.x = pack2_bf16_rne(v.x, v.y);
      p.y = pack2_bf16_rne(v.z, v.w);
      *(uint2*)&As[buf][r][cc] = p;
    }
    for (int idx = tid * 4; idx < TN * TK; idx += 256 * 4) {
      const int r = idx / TK, cc = idx % TK;
      const int gn = tileN + r, gk = k0 + cc;
      uint2 p;
      if (gn < N) {
        const float4 v = *(const float4*)(W + (size_t)gn * (size_t)K + gk);
        p.x = pack2_bf16_rne(v.x, v.y);
        p.y = pack2_bf16_rne(v.z, v.w);
      } else {
        p.x = 0u; p.y = 0u;
      }
      *(uint2*)&Bs[buf][r][cc] = p;
    }
  };

  stage(0, 0);

  int cur = 0;
  for (int k0 = 0; k0 < K; k0 += TK, cur ^= 1) {
    __syncthreads();                     // staged tile visible; prev reads done
    if (k0 + TK < K) stage(k0 + TK, cur ^ 1);

    // prefetch the tile after next (global_prefetch_b8)
    if (k0 + 2 * TK < K) {
      const int gr = tileM + (tid & 127);
      const int gc = k0 + 2 * TK;
      const float* p = (gc < ksplit)
          ? (X0 + (size_t)gr * (size_t)ksplit + gc)
          : (X1 + (size_t)gr * (size_t)(K - ksplit) + (gc - ksplit));
      __builtin_prefetch(p, 0, 1);
    }

    // ---- load ALL fragments first, then a back-to-back WMMA burst ----
    // A fragment (16x32 bf16): lanes 0-15 -> M=lane, K 0-7 & 16-23;
    // lanes 16-31 -> M=lane-16, K 8-15 & 24-31 (ISA 16-bit A layout).
    const int ak = (lane < 16) ? 0 : 8;
    v16bf af[2];
    #pragma unroll
    for (int a = 0; a < 2; ++a) {
      const int mrow = wm * 32 + a * 16 + (lane & 15);
      const uint4 a0 = *(const uint4*)&As[cur][mrow][ak];
      const uint4 a1 = *(const uint4*)&As[cur][mrow][ak + 16];
      const v8u raw = { a0.x, a0.y, a0.z, a0.w, a1.x, a1.y, a1.z, a1.w };
      af[a] = __builtin_bit_cast(v16bf, raw);
    }
    // B fragment (32x16): lanes 0-15 -> N=lane, K 0-7 & 16-23; lanes 16-31 -> K 8-15 & 24-31
    const int bk = (lane < 16) ? 0 : 8;
    v16bf bf[4];
    #pragma unroll
    for (int nt = 0; nt < 4; ++nt) {
      const int nrow = wn * 64 + nt * 16 + (lane & 15);
      const uint4 b0 = *(const uint4*)&Bs[cur][nrow][bk];
      const uint4 b1 = *(const uint4*)&Bs[cur][nrow][bk + 16];
      const v8u raw = { b0.x, b0.y, b0.z, b0.w, b1.x, b1.y, b1.z, b1.w };
      bf[nt] = __builtin_bit_cast(v16bf, raw);
    }
    #pragma unroll
    for (int a = 0; a < 2; ++a)
      #pragma unroll
      for (int nt = 0; nt < 4; ++nt)
        acc[a][nt] = __builtin_amdgcn_wmma_f32_16x16x32_bf16(
            false, af[a], false, bf[nt], (short)0, acc[a][nt], false, false);
  }

  // epilogue: C/D layout -> VGPR r holds row r (lanes 0-15) / row r+8 (16-31)
  #pragma unroll
  for (int a = 0; a < 2; ++a) {
    const int rbase = tileM + wm * 32 + a * 16 + ((lane < 16) ? 0 : 8);
    #pragma unroll
    for (int nt = 0; nt < 4; ++nt) {
      const int col = tileN + wn * 64 + nt * 16 + (lane & 15);
      if (col < N) {
        const float bv = bias[col];
        #pragma unroll
        for (int r = 0; r < 8; ++r) {
          float v = acc[a][nt][r] + bv;
          if (relu) v = fmaxf(v, 0.0f);
          Y[(size_t)(rbase + r) * (size_t)N + col] = v;
        }
      }
    }
  }
}

// ---------------------------------------------------------------------------
// Qinv = Qc Qc^T, Q0inv = Q0c Q0c^T, AQinv = A^T Qinv, AQinvA = AQinv A
// ---------------------------------------------------------------------------
__global__ __launch_bounds__(1024) void prep_mats_kernel(
    const float* __restrict__ A, const float* __restrict__ QinvChol,
    const float* __restrict__ Q0invChol,
    float* __restrict__ Qinv, float* __restrict__ Q0inv,
    float* __restrict__ AQinv, float* __restrict__ AQinvA)
{
  __shared__ float Am[DZN][DZN], Qc[DZN][DZN], Q0c[DZN][DZN];
  __shared__ float Qi[DZN][DZN], AQ[DZN][DZN];
  const int tid = threadIdx.x;
  const int i = tid >> 5, j = tid & 31;
  Am[i][j]  = A[tid];
  Qc[i][j]  = QinvChol[tid];
  Q0c[i][j] = Q0invChol[tid];
  __syncthreads();
  float s = 0.f, s0 = 0.f;
  for (int k = 0; k < DZN; ++k) { s += Qc[i][k]*Qc[j][k]; s0 += Q0c[i][k]*Q0c[j][k]; }
  Qi[i][j] = s; Qinv[tid] = s; Q0inv[tid] = s0;
  __syncthreads();
  float t = 0.f;
  for (int k = 0; k < DZN; ++k) t += Am[k][i] * Qi[k][j];
  AQ[i][j] = t; AQinv[tid] = t;
  __syncthreads();
  float u = 0.f;
  for (int k = 0; k < DZN; ++k) u += AQ[i][k] * Am[k][j];
  AQinvA[tid] = u;
}

// ---------------------------------------------------------------------------
// Per step t: Lambda = Lc Lc^T ; AA = Lambda + diag_add(t) ; lambdaMu = Lambda @ mu
// ---------------------------------------------------------------------------
__global__ __launch_bounds__(1024) void lambda_kernel(
    const float* __restrict__ covOut, const float* __restrict__ meanOut,
    const float* __restrict__ Qinv, const float* __restrict__ Q0inv,
    const float* __restrict__ AQinvA,
    float* __restrict__ AAbuf, float* __restrict__ lambdaMu)
{
  __shared__ float Lc[DZN][DZN], Lam[DZN][DZN], mrow[DZN];
  const int t = blockIdx.x;
  const int tid = threadIdx.x;
  const int i = tid >> 5, j = tid & 31;
  Lc[i][j] = covOut[(size_t)t * (DZN*DZN) + tid];
  if (tid < DZN) mrow[tid] = meanOut[(size_t)t * DZN + tid];
  __syncthreads();
  float s = 0.f;
  for (int k = 0; k < DZN; ++k) s += Lc[i][k] * Lc[j][k];
  Lam[i][j] = s;
  float d;
  if (t == 0)                 d = Q0inv[tid] + AQinvA[tid];
  else if (t == T_STEPS - 1)  d = Qinv[tid];
  else                        d = Qinv[tid] + AQinvA[tid];
  AAbuf[(size_t)t * (DZN*DZN) + tid] = s + d;
  __syncthreads();
  if (tid < DZN) {
    float lm = 0.f;
    for (int k = 0; k < DZN; ++k) lm += Lam[tid][k] * mrow[k];
    lambdaMu[(size_t)t * DZN + tid] = lm;
  }
}

// ---------------------------------------------------------------------------
// Sequential block-tridiagonal Cholesky: single workgroup, LDS-resident blocks.
// R[t] = lower chol blocks, C[t-1] = B^T L^{-T}, ln_det accumulated.
// ---------------------------------------------------------------------------
__global__ __launch_bounds__(1024) void chol_scan_kernel(
    const float* __restrict__ AAbuf, const float* __restrict__ AQinv,
    float* __restrict__ Rbuf, float* __restrict__ Cbuf,
    float* __restrict__ lnDetOut)
{
  __shared__ float Lp[DZN][DZN], Mm[DZN][DZN], Yc[DZN][DZN], Bc[DZN][DZN];
  __shared__ float red[DZN];
  const int tid = threadIdx.x;
  const int i = tid >> 5, j = tid & 31;
  Bc[i][j] = -AQinv[tid];        // BB[t] = -AQinv (constant over t)
  float acc = 0.f;
  __syncthreads();
  for (int t = 0; t < T_STEPS; ++t) {
    Mm[i][j] = AAbuf[(size_t)t * (DZN*DZN) + tid];
    __syncthreads();
    if (t > 0) {
      // forward-substitute Lp * Y = Bc, one column per thread (32 active)
      if (tid < DZN) {
        const int col = tid;
        for (int r = 0; r < DZN; ++r) {
          float s = Bc[r][col];
          for (int k = 0; k < r; ++k) s -= Lp[r][k] * Yc[k][col];
          Yc[r][col] = s / Lp[r][r];
        }
      }
      __syncthreads();
      Cbuf[(size_t)(t - 1) * (DZN*DZN) + tid] = Yc[j][i];  // C = Y^T
      float s = 0.f;                                       // Mm -= C C^T
      for (int k = 0; k < DZN; ++k) s += Yc[k][i] * Yc[k][j];
      Mm[i][j] -= s;
      __syncthreads();
    }
    // in-place lower Cholesky of Mm
    for (int p = 0; p < DZN; ++p) {
      if (tid == 0) Mm[p][p] = sqrtf(Mm[p][p]);
      __syncthreads();
      if (tid > p && tid < DZN) Mm[tid][p] /= Mm[p][p];
      __syncthreads();
      if (j > p && i >= j) Mm[i][j] -= Mm[i][p] * Mm[j][p];
      __syncthreads();
    }
    Rbuf[(size_t)t * (DZN*DZN) + tid] = (j <= i) ? Mm[i][j] : 0.f;
    if (tid < DZN) acc += logf(Mm[tid][tid]);
    Lp[i][j] = (j <= i) ? Mm[i][j] : 0.f;
    __syncthreads();
  }
  if (tid < DZN) red[tid] = acc;
  __syncthreads();
  if (tid == 0) {
    float s = 0.f;
    for (int k = 0; k < DZN; ++k) s += red[k];
    *lnDetOut = -2.0f * s;
  }
}

// ---------------------------------------------------------------------------
// Forward bidiagonal solve: x0 = R0^{-1} b0 ; x_t = R_t^{-1}(b_t - C_{t-1} x_{t-1})
// ---------------------------------------------------------------------------
__global__ __launch_bounds__(256) void fwd_solve_kernel(
    const float* __restrict__ Rbuf, const float* __restrict__ Cbuf,
    const float* __restrict__ b, float* __restrict__ xout)
{
  __shared__ float L[DZN][DZN], Cc[DZN][DZN], rhs[DZN], xc[DZN], xp[DZN];
  const int tid = threadIdx.x;
  for (int t = 0; t < T_STEPS; ++t) {
    __syncthreads();
    for (int idx = tid; idx < DZN*DZN; idx += 256) {
      L[idx >> 5][idx & 31] = Rbuf[(size_t)t * (DZN*DZN) + idx];
      if (t > 0) Cc[idx >> 5][idx & 31] = Cbuf[(size_t)(t - 1) * (DZN*DZN) + idx];
    }
    if (tid < DZN) rhs[tid] = b[(size_t)t * DZN + tid];
    __syncthreads();
    if (t > 0 && tid < DZN) {
      float s = 0.f;
      for (int k = 0; k < DZN; ++k) s += Cc[tid][k] * xp[k];
      rhs[tid] -= s;
    }
    __syncthreads();
    for (int p = 0; p < DZN; ++p) {
      if (tid == 0) xc[p] = rhs[p] / L[p][p];
      __syncthreads();
      if (tid > p && tid < DZN) rhs[tid] -= L[tid][p] * xc[p];
      __syncthreads();
    }
    if (tid < DZN) { xout[(size_t)t * DZN + tid] = xc[tid]; xp[tid] = xc[tid]; }
  }
}

// ---------------------------------------------------------------------------
// Backward solve with U = R^T: x_{T-1} = R^{-T} b ; x_t = R_t^{-T}(b_t - C_t^T x_{t+1})
// ---------------------------------------------------------------------------
__global__ __launch_bounds__(256) void bwd_solve_kernel(
    const float* __restrict__ Rbuf, const float* __restrict__ Cbuf,
    const float* __restrict__ b, float* __restrict__ xout)
{
  __shared__ float L[DZN][DZN], Cc[DZN][DZN], rhs[DZN], xc[DZN], xn[DZN];
  const int tid = threadIdx.x;
  for (int tt = 0; tt < T_STEPS; ++tt) {
    const int t = T_STEPS - 1 - tt;
    __syncthreads();
    for (int idx = tid; idx < DZN*DZN; idx += 256) {
      L[idx >> 5][idx & 31] = Rbuf[(size_t)t * (DZN*DZN) + idx];
      if (t < T_STEPS - 1) Cc[idx >> 5][idx & 31] = Cbuf[(size_t)t * (DZN*DZN) + idx];
    }
    if (tid < DZN) rhs[tid] = b[(size_t)t * DZN + tid];
    __syncthreads();
    if (t < T_STEPS - 1 && tid < DZN) {
      float s = 0.f;
      for (int k = 0; k < DZN; ++k) s += Cc[k][tid] * xn[k];   // C^T x
      rhs[tid] -= s;
    }
    __syncthreads();
    for (int pp = 0; pp < DZN; ++pp) {
      const int p = DZN - 1 - pp;
      if (tid == 0) xc[p] = rhs[p] / L[p][p];
      __syncthreads();
      if (tid < p) rhs[tid] -= L[p][tid] * xc[p];              // U[k][p] = L[p][k]
      __syncthreads();
    }
    if (tid < DZN) { xout[(size_t)t * DZN + tid] = xc[tid]; xn[tid] = xc[tid]; }
  }
}

// ---------------------------------------------------------------------------
extern "C" void kernel_launch(void* const* d_in, const int* in_sizes, int n_in,
                              void* d_out, int out_size, void* d_ws, size_t ws_size,
                              hipStream_t stream) {
  (void)in_sizes; (void)out_size; (void)ws_size;
  const float* in[64];
  for (int i = 0; i < n_in && i < 64; ++i) in[i] = (const float*)d_in[i];

  // dict-insertion-order flattening:
  // [0..15]  mean branch: (W,b) x {num_in,num_h1,num_h3,cat_in,cat_h1,concat,h1,out}
  // [16..31] cov  branch: same
  // [32] A, [33] QinvChol, [34] Q0invChol, [35] x, [36] c
  const float* A         = in[32];
  const float* QinvChol  = in[33];
  const float* Q0invChol = in[34];
  const float* x         = in[35];
  const float* c         = in[36];

  float* ws = (float*)d_ws;
  const size_t TH = (size_t)T_STEPS * DHN;
  float* buf0  = ws;                 // T x 1024 scratch / R
  float* buf1  = ws + TH;            // T x 1024 scratch / AA
  float* buf2  = ws + 2 * TH;        // T x 1024 scratch / C
  float* covO  = ws + 3 * TH;        // T x 1024 cov encoder output
  float* meanO = ws + 4 * TH;        // T x 32
  float* lmu   = meanO + (size_t)T_STEPS * DZN;
  float* ib    = lmu   + (size_t)T_STEPS * DZN;
  float* Qi    = ib    + (size_t)T_STEPS * DZN;
  float* Q0i   = Qi  + DZN * DZN;
  float* AQ    = Q0i + DZN * DZN;
  float* AQA   = AQ  + DZN * DZN;

  float* outMu = (float*)d_out;
  float* lnDet = outMu + (size_t)T_STEPS * DZN;

  const dim3 blk(256);
  auto gemm = [&](const float* X0, const float* X1, int ksplit,
                  const float* W, const float* bias, float* Y,
                  int N, int K, int relu) {
    dim3 grid((N + TN - 1) / TN, T_STEPS / TM);
    hipLaunchKernelGGL(gemm_bias_act_wmma, grid, blk, 0, stream,
                       X0, X1, ksplit, W, bias, Y, T_STEPS, N, K, relu);
  };

  for (int br = 0; br < 2; ++br) {
    const int B = (br == 0) ? 0 : 16;
    float* outBuf = (br == 0) ? meanO : covO;
    const int Nout = (br == 0) ? DZN : DZN * DZN;
    gemm(x,    x,    DXN, in[B+0],  in[B+1],  buf0,   DHN,  DXN,     1); // num_in
    gemm(buf0, buf0, DHN, in[B+2],  in[B+3],  buf1,   DHN,  DHN,     1); // num_h1
    gemm(buf1, buf1, DHN, in[B+4],  in[B+5],  buf0,   DHN,  DHN,     1); // num_h3
    gemm(c,    c,    DCN, in[B+6],  in[B+7],  buf1,   DHN,  DCN,     1); // cat_in
    gemm(buf1, buf1, DHN, in[B+8],  in[B+9],  buf2,   DHN,  DHN,     1); // cat_h1
    gemm(buf0, buf2, DHN, in[B+10], in[B+11], buf1,   DHN,  2*DHN,   1); // concat (split-K)
    gemm(buf1, buf1, DHN, in[B+12], in[B+13], buf0,   DHN,  DHN,     1); // h1
    gemm(buf0, buf0, DHN, in[B+14], in[B+15], outBuf, Nout, DHN,     0); // out
  }

  hipLaunchKernelGGL(prep_mats_kernel, dim3(1), dim3(1024), 0, stream,
                     A, QinvChol, Q0invChol, Qi, Q0i, AQ, AQA);
  hipLaunchKernelGGL(lambda_kernel, dim3(T_STEPS), dim3(1024), 0, stream,
                     covO, meanO, Qi, Q0i, AQA, buf1, lmu);
  hipLaunchKernelGGL(chol_scan_kernel, dim3(1), dim3(1024), 0, stream,
                     buf1, AQ, buf0, buf2, lnDet);
  hipLaunchKernelGGL(fwd_solve_kernel, dim3(1), dim3(256), 0, stream,
                     buf0, buf2, lmu, ib);
  hipLaunchKernelGGL(bwd_solve_kernel, dim3(1), dim3(256), 0, stream,
                     buf0, buf2, ib, outMu);
}
